// LinearAttention_31473520345438
// MI455X (gfx1250) — matrix operand order repaired and checked by reading, not compiled, not run on hardware
//
#include <hip/hip_runtime.h>
#include <hip/hip_bf16.h>
#include <stdint.h>

// ---------------------------------------------------------------------------
// Problem constants (reference: B=4, S=4096, D=UNITS=NRF=1024)
// ---------------------------------------------------------------------------
#define BATCH   4
#define SEQ     4096
#define DIM     1024            // D == UNITS == NRF
#define MTOT    (BATCH * SEQ)   // 16384 rows for the fused [B*S, D] GEMMs

typedef __attribute__((ext_vector_type(16))) __bf16 v16bf;
typedef __attribute__((ext_vector_type(8)))  float  v8f;
typedef __attribute__((ext_vector_type(4)))  unsigned int v4u;
typedef __attribute__((ext_vector_type(4)))  int v4i;
typedef unsigned short u16;

union Frag16 {                      // 16 bf16 = one WMMA A/B operand
    v16bf v;
    v4u   q[2];                     // two 16-byte chunks
    u16   u[16];
};

__device__ __forceinline__ u16 f2bf(float f) {
    unsigned int x = __float_as_uint(f);
    unsigned int r = x + 0x7fffu + ((x >> 16) & 1u);   // round-to-nearest-even
    return (u16)(r >> 16);
}

__device__ __forceinline__ v8f wmma_bf16(const Frag16& a, const Frag16& b, v8f c) {
    return __builtin_amdgcn_wmma_f32_16x16x32_bf16(
        /*neg_a=*/false, a.v, /*neg_b=*/false, b.v,
        /*c_mod=*/(short)0, c, /*reuse_a=*/false, /*reuse_b=*/false);
}

// ---------------------------------------------------------------------------
// Async global -> LDS 16B copy (CDNA5 GLOBAL_LOAD_ASYNC_TO_LDS_B128, ASYNCcnt)
// with a compile-safe synchronous fallback if the builtin isn't declared.
// Probe-learned signature: (v4i AS1* gsrc, v4i AS3* ldst, imm offset, imm cpol)
// ---------------------------------------------------------------------------
#if defined(__AMDGCN__) && __has_builtin(__builtin_amdgcn_global_load_async_to_lds_b128)
#define ASYNC_LDS 1
typedef __attribute__((address_space(1))) v4i GV4;   // global-AS 16B vector
typedef __attribute__((address_space(3))) v4i LV4;   // LDS-AS 16B vector
#else
#define ASYNC_LDS 0
#endif

__device__ __forceinline__ void tile_copy16(u16* ldst, const u16* gsrc) {
#if ASYNC_LDS
    __builtin_amdgcn_global_load_async_to_lds_b128(
        (GV4*)gsrc, (LV4*)ldst, /*offset=*/0, /*cpol=*/0);
#else
    *(v4u*)ldst = *(const v4u*)gsrc;
#endif
}

__device__ __forceinline__ void tile_copy_wait() {
#if ASYNC_LDS
#if __has_builtin(__builtin_amdgcn_s_wait_asynccnt)
    __builtin_amdgcn_s_wait_asynccnt(0);
#else
    asm volatile("s_wait_asynccnt 0" ::: "memory");
#endif
#endif
}

// ---------------------------------------------------------------------------
// f32 -> bf16 conversion (row-major copy)
// ---------------------------------------------------------------------------
__global__ void cvt_f32_bf16(const float* __restrict__ in, u16* __restrict__ out, int n) {
    int i = blockIdx.x * 256 + threadIdx.x;
    if (i < n) out[i] = f2bf(in[i]);
}

// f32 [K][N] -> bf16 transposed [N][K]  (weights: done once, L2-resident after)
__global__ void cvt_t_f32_bf16(const float* __restrict__ in, u16* __restrict__ out,
                               int K, int N) {
    int i = blockIdx.x * 256 + threadIdx.x;
    if (i < K * N) {
        int k = i / N, n = i - k * N;
        out[(size_t)n * K + k] = f2bf(in[i]);
    }
}

// ---------------------------------------------------------------------------
// Tiled WMMA GEMM:  Out = epi( A[M,K](bf16) @ B + bias ),  B given TRANSPOSED
// as BT[N,K] so B-fragments are contiguous 16B loads with k = 16*g + s, as for A.
//   block tile 128x128, 256 threads = 8 waves (wave: 32 rows x 64 cols),
//   K-step 32, double-buffered async LDS staging of the A tile only.
//   EPI:    0 = +bias, 1 = cos(+bias)
//   STORET: 0 = row-major [M,N]; 1 = per-batch transposed [b][N][SEQ] (V^T)
// ---------------------------------------------------------------------------
template <int EPI, int STORET>
__global__ __launch_bounds__(256)
void wmma_gemm_bf16_k(const u16* __restrict__ A, const u16* __restrict__ BT,
                      const float* __restrict__ bias, u16* __restrict__ Out,
                      int Ktot, int Ntot) {
    __shared__ u16 sA[2][128 * 40];   // [row][k] stride 40 (bank-conflict pad)

    const int tid  = threadIdx.x;
    const int wave = tid >> 5;
    const int lane = tid & 31;
    const int g    = lane >> 4;
    const int l15  = lane & 15;
    const int wr   = wave & 3;        // row-group: 32 rows
    const int wc   = wave >> 2;       // col-group: 64 cols

    const int rowBase = blockIdx.y * 128;
    const int colBase = blockIdx.x * 128;

    v8f acc[2][4];
#pragma unroll
    for (int mt = 0; mt < 2; ++mt)
#pragma unroll
        for (int nt = 0; nt < 4; ++nt)
#pragma unroll
            for (int i = 0; i < 8; ++i) acc[mt][nt][i] = 0.0f;

    // A-tile stage: 128x32 bf16 = 512 x 16B chunks, 2 per thread
    auto loadA = [&](int buf, int k0) {
#pragma unroll
        for (int p = 0; p < 2; ++p) {
            const int c   = p * 256 + tid;
            const int r   = c >> 2;
            const int off = (c & 3) * 8;
            tile_copy16(&sA[buf][r * 40 + off],
                        A + (size_t)(rowBase + r) * Ktot + k0 + off);
        }
    };

    loadA(0, 0);
    tile_copy_wait();
    __syncthreads();

    const int ksteps = Ktot >> 5;
    for (int kk = 0; kk < ksteps; ++kk) {
        const int cur = kk & 1;
        if (kk + 1 < ksteps) loadA(cur ^ 1, (kk + 1) << 5);  // prefetch next tile

        Frag16 a[2];
#pragma unroll
        for (int mt = 0; mt < 2; ++mt) {
            const u16* ap = &sA[cur][(wr * 32 + mt * 16 + l15) * 40];
            a[mt].q[0] = *(const v4u*)(ap + 16 * g);         // K = 16g..16g+7
            a[mt].q[1] = *(const v4u*)(ap + 16 * g + 8);     // K = 16g+8..
        }
        const int k0 = kk << 5;
#pragma unroll
        for (int nt = 0; nt < 4; ++nt) {
            const int n   = colBase + wc * 64 + nt * 16 + l15;
            const u16* bp = BT + (size_t)n * Ktot + k0 + 16 * g;  // contiguous K
            Frag16 b;
            b.q[0] = *(const v4u*)bp;
            b.q[1] = *(const v4u*)(bp + 8);
#pragma unroll
            for (int mt = 0; mt < 2; ++mt)
                acc[mt][nt] = wmma_bf16(a[mt], b, acc[mt][nt]);
        }
        tile_copy_wait();
        __syncthreads();
    }

    // epilogue: C tile mapping — VGPR r -> row 8g+r, col = lane&15
#pragma unroll
    for (int mt = 0; mt < 2; ++mt)
#pragma unroll
        for (int nt = 0; nt < 4; ++nt) {
            const int col = colBase + wc * 64 + nt * 16 + l15;
            const float bv = bias[col];
#pragma unroll
            for (int r = 0; r < 8; ++r) {
                const int row = rowBase + wr * 32 + mt * 16 + g * 8 + r;
                float v = acc[mt][nt][r] + bv;
                if (EPI == 1) v = __cosf(v);
                const u16 h = f2bf(v);
                if (STORET == 0) {
                    Out[(size_t)row * Ntot + col] = h;
                } else {             // [b][N][SEQ]  (V^T for the PV stage)
                    const int bb = row >> 12;        // row / SEQ
                    const int s  = row & (SEQ - 1);  // row % SEQ
                    Out[((size_t)bb * Ntot + col) * SEQ + s] = h;
                }
            }
        }
}

// ---------------------------------------------------------------------------
// Flash attention over feature-mapped Q/K with V^T:
//   per block: 16 query rows, full 1024 output cols, online softmax over 4096
//   keys in chunks of 512. 256 threads = 8 waves; wave w owns output cols
//   [w*128, w*128+128) and score keys [chunk + w*64, +64).
// Dynamic LDS layout (bytes):
//   sQ  [16][1024] bf16 : 32768
//   sS  [16][512]  f32  : 32768
//   sP  [16][512]  bf16 : 16384
//   stats: m[16], l[16], scale[16], part[16][16] f32
// ---------------------------------------------------------------------------
#define ATT_LDS_BYTES (32768 + 32768 + 16384 + (16 + 16 + 16 + 256) * 4)

__global__ __launch_bounds__(256)
void flash_attn_k(const u16* __restrict__ Qh, const u16* __restrict__ Kh,
                  const u16* __restrict__ VT, float* __restrict__ Out) {
    extern __shared__ char smem[];
    u16*   sQ     = (u16*)smem;                       // 16 x 1024
    float* sS     = (float*)(smem + 32768);           // 16 x 512
    u16*   sP     = (u16*)(smem + 65536);             // 16 x 512
    float* sM     = (float*)(smem + 81920);
    float* sL     = sM + 16;
    float* sScale = sM + 32;
    float* sPart  = sM + 48;                          // 16 x 16

    const int tid  = threadIdx.x;
    const int wave = tid >> 5;
    const int lane = tid & 31;
    const int g    = lane >> 4;
    const int l15  = lane & 15;

    const int b     = blockIdx.x >> 8;                // 256 query-tiles per batch
    const int qRow0 = (blockIdx.x & 255) * 16;

    const u16* Qb = Qh + ((size_t)b * SEQ + qRow0) * DIM;
    const u16* Kb = Kh + (size_t)b * SEQ * DIM;
    const u16* Vb = VT + (size_t)b * DIM * SEQ;       // [d][s]
    float*     Ob = Out + ((size_t)b * SEQ + qRow0) * DIM;

    if (tid < 16) { sM[tid] = -3.4e38f; sL[tid] = 0.0f; }

    // stage the 16 x 1024 Q_hat tile (async; reused for every key chunk)
    for (int i = tid; i < 2048; i += 256)
        tile_copy16(sQ + i * 8, Qb + i * 8);
    tile_copy_wait();
    __syncthreads();

    v8f acc[8];
#pragma unroll
    for (int nt = 0; nt < 8; ++nt)
#pragma unroll
        for (int i = 0; i < 8; ++i) acc[nt][i] = 0.0f;

    const int rrow = tid >> 4;      // softmax helper indexing: 16 threads/row
    const int rseg = tid & 15;

    for (int ch = 0; ch < SEQ / 512; ++ch) {
        const int chunk0 = ch * 512;

        // ---- phase 1: scores S = Qhat @ Khat^T for this wave's 64 keys ----
        v8f sc[4];
#pragma unroll
        for (int nt = 0; nt < 4; ++nt)
#pragma unroll
            for (int i = 0; i < 8; ++i) sc[nt][i] = 0.0f;

        const int keyBase = chunk0 + wave * 64;
        for (int fs = 0; fs < DIM / 32; ++fs) {
            Frag16 a;
            a.q[0] = *(const v4u*)(sQ + l15 * DIM + fs * 32 + 16 * g);
            a.q[1] = *(const v4u*)(sQ + l15 * DIM + fs * 32 + 16 * g + 8);
#pragma unroll
            for (int nt = 0; nt < 4; ++nt) {
                const int key = keyBase + nt * 16 + l15;
                const u16* kp = Kb + (size_t)key * DIM + fs * 32 + 16 * g;
                Frag16 bf;
                bf.q[0] = *(const v4u*)kp;          // B = Khat^T: contiguous in F
                bf.q[1] = *(const v4u*)(kp + 8);
                sc[nt] = wmma_bf16(a, bf, sc[nt]);
            }
        }
#pragma unroll
        for (int nt = 0; nt < 4; ++nt)
#pragma unroll
            for (int r = 0; r < 8; ++r)
                sS[(g * 8 + r) * 512 + wave * 64 + nt * 16 + l15] = sc[nt][r];
        __syncthreads();

        // ---- phase 2: chunk row-max -> online max update ----
        {
            float mx = -3.4e38f;
            const float* p = sS + rrow * 512 + rseg * 32;
#pragma unroll
            for (int j = 0; j < 32; ++j) mx = fmaxf(mx, p[j]);
            sPart[rrow * 16 + rseg] = mx;
        }
        __syncthreads();
        if (tid < 16) {
            float m = sM[tid];
            const float mo = m;
#pragma unroll
            for (int j = 0; j < 16; ++j) m = fmaxf(m, sPart[tid * 16 + j]);
            sScale[tid] = __expf(mo - m);           // first chunk: exp(-inf) = 0
            sM[tid] = m;
        }
        __syncthreads();

        // ---- phase 3: P = exp(S - m) (bf16) + row-sum update ----
        {
            const float mrow = sM[rrow];
            float sum = 0.0f;
            float* p  = sS + rrow * 512 + rseg * 32;
            u16*   pp = sP + rrow * 512 + rseg * 32;
#pragma unroll
            for (int j = 0; j < 32; ++j) {
                const float e = __expf(p[j] - mrow);
                pp[j] = f2bf(e);
                sum += e;
            }
            sPart[rrow * 16 + rseg] = sum;
        }
        __syncthreads();
        if (tid < 16) {
            float s = 0.0f;
#pragma unroll
            for (int j = 0; j < 16; ++j) s += sPart[tid * 16 + j];
            sL[tid] = sL[tid] * sScale[tid] + s;
        }
        __syncthreads();

        // ---- phase 4: rescale accumulators, then acc += P @ V ----
        float scl[8];
#pragma unroll
        for (int r = 0; r < 8; ++r) scl[r] = sScale[g * 8 + r];
#pragma unroll
        for (int nt = 0; nt < 8; ++nt)
#pragma unroll
            for (int r = 0; r < 8; ++r) acc[nt][r] *= scl[r];

        for (int kt = 0; kt < 16; ++kt) {           // 512 keys in steps of 32
            Frag16 a;
            a.q[0] = *(const v4u*)(sP + l15 * 512 + kt * 32 + 16 * g);
            a.q[1] = *(const v4u*)(sP + l15 * 512 + kt * 32 + 16 * g + 8);
            const int key0 = chunk0 + kt * 32 + 16 * g;
#pragma unroll
            for (int nt = 0; nt < 8; ++nt) {
                const int n = wave * 128 + nt * 16 + l15;
                const u16* vp = Vb + (size_t)n * SEQ + key0;   // V^T: contiguous keys
                Frag16 bf;
                bf.q[0] = *(const v4u*)vp;
                bf.q[1] = *(const v4u*)(vp + 8);
                acc[nt] = wmma_bf16(a, bf, acc[nt]);
            }
        }
        __syncthreads();
    }

    // ---- final: out = acc / l ----
    float invl[8];
#pragma unroll
    for (int r = 0; r < 8; ++r) invl[r] = 1.0f / sL[g * 8 + r];
#pragma unroll
    for (int nt = 0; nt < 8; ++nt)
#pragma unroll
        for (int r = 0; r < 8; ++r)
            Ob[(size_t)(g * 8 + r) * DIM + wave * 128 + nt * 16 + l15] =
                acc[nt][r] * invl[r];
}

// ---------------------------------------------------------------------------
// host-side orchestration
// ---------------------------------------------------------------------------
extern "C" void kernel_launch(void* const* d_in, const int* in_sizes, int n_in,
                              void* d_out, int out_size, void* d_ws, size_t ws_size,
                              hipStream_t stream) {
    const float* query = (const float*)d_in[0];
    const float* value = (const float*)d_in[1];
    const float* Wq    = (const float*)d_in[2];
    const float* bq    = (const float*)d_in[3];
    const float* Wk    = (const float*)d_in[4];
    const float* bk    = (const float*)d_in[5];
    const float* Wv    = (const float*)d_in[6];
    const float* bv    = (const float*)d_in[7];
    const float* Wr    = (const float*)d_in[8];
    const float* br    = (const float*)d_in[9];
    float* out = (float*)d_out;

    // workspace layout (bf16 = u16), ~232 MB total
    char* ws = (char*)d_ws;
    const size_t SZ_BIG = (size_t)MTOT * DIM * sizeof(u16);  // 32 MiB
    const size_t SZ_W   = (size_t)DIM * DIM * sizeof(u16);   // 2 MiB
    u16* qbf  = (u16*)(ws);
    u16* vbf  = (u16*)(ws + SZ_BIG);
    u16* WqT  = (u16*)(ws + 2 * SZ_BIG);                 // transposed bf16 weights
    u16* WkT  = (u16*)(ws + 2 * SZ_BIG + SZ_W);
    u16* WvT  = (u16*)(ws + 2 * SZ_BIG + 2 * SZ_W);
    u16* WrT  = (u16*)(ws + 2 * SZ_BIG + 3 * SZ_W);
    u16* Qtmp = (u16*)(ws + 2 * SZ_BIG + 4 * SZ_W);
    u16* Ktmp = (u16*)(ws + 3 * SZ_BIG + 4 * SZ_W);
    u16* VT   = (u16*)(ws + 4 * SZ_BIG + 4 * SZ_W);
    u16* Qhat = (u16*)(ws + 5 * SZ_BIG + 4 * SZ_W);
    u16* Khat = (u16*)(ws + 6 * SZ_BIG + 4 * SZ_W);

    const int nBig = MTOT * DIM;          // 16777216
    const int nW   = DIM * DIM;           // 1048576
    cvt_f32_bf16<<<(nBig + 255) / 256, 256, 0, stream>>>(query, qbf, nBig);
    cvt_f32_bf16<<<(nBig + 255) / 256, 256, 0, stream>>>(value, vbf, nBig);
    cvt_t_f32_bf16<<<(nW + 255) / 256, 256, 0, stream>>>(Wq, WqT, DIM, DIM);
    cvt_t_f32_bf16<<<(nW + 255) / 256, 256, 0, stream>>>(Wk, WkT, DIM, DIM);
    cvt_t_f32_bf16<<<(nW + 255) / 256, 256, 0, stream>>>(Wv, WvT, DIM, DIM);
    cvt_t_f32_bf16<<<(nW + 255) / 256, 256, 0, stream>>>(Wr, WrT, DIM, DIM);

    dim3 gGemm(DIM / 128, MTOT / 128);    // 8 x 128
    dim3 bGemm(256);
    // projections
    wmma_gemm_bf16_k<0, 0><<<gGemm, bGemm, 0, stream>>>(qbf, WqT, bq, Qtmp, DIM, DIM);
    wmma_gemm_bf16_k<0, 0><<<gGemm, bGemm, 0, stream>>>(vbf, WkT, bk, Ktmp, DIM, DIM);
    wmma_gemm_bf16_k<0, 1><<<gGemm, bGemm, 0, stream>>>(vbf, WvT, bv, VT,   DIM, DIM);
    // random-feature maps: cos(X @ Wr + br)
    wmma_gemm_bf16_k<1, 0><<<gGemm, bGemm, 0, stream>>>(Qtmp, WrT, br, Qhat, DIM, DIM);
    wmma_gemm_bf16_k<1, 0><<<gGemm, bGemm, 0, stream>>>(Ktmp, WrT, br, Khat, DIM, DIM);

    // flash attention: 4 batches x 256 query tiles
    flash_attn_k<<<dim3(BATCH * (SEQ / 16)), dim3(256), ATT_LDS_BYTES, stream>>>(
        Qhat, Khat, VT, out);
}
